// new_psp2Head_86423331930285
// MI455X (gfx1250) — matrix operand order repaired and checked by reading, not compiled, not run on hardware
//
#include <hip/hip_runtime.h>

typedef __attribute__((ext_vector_type(16))) _Float16 v16h;
typedef __attribute__((ext_vector_type(8)))  _Float16 v8h;
typedef __attribute__((ext_vector_type(8)))  float    v8f;

#define CIN   2048
#define CI    512
#define HW    4096
#define NB    2
#define NCELL 194
#define NCLS  19

// ---------------------------------------------------------------- fold BN
__global__ void fold_ab_kernel(const float* __restrict__ bn_s, const float* __restrict__ bn_b,
                               const float* __restrict__ bn_m, const float* __restrict__ bn_v,
                               const float* __restrict__ s52, const float* __restrict__ b52,
                               const float* __restrict__ m52, const float* __restrict__ v52,
                               float* __restrict__ alpha, float* __restrict__ beta) {
  int t = blockIdx.x * blockDim.x + threadIdx.x;
  if (t >= 8 * CI) return;
  int i = t / CI, o = t % CI;
  float s, b, m, v;
  if (i < 7) { s = bn_s[i*CI+o]; b = bn_b[i*CI+o]; m = bn_m[i*CI+o]; v = bn_v[i*CI+o]; }
  else       { s = s52[o];       b = b52[o];       m = m52[o];       v = v52[o];       }
  float a = s * rsqrtf(v + 1e-5f);
  alpha[t] = a;
  beta[t]  = b - m * a;
}

// ---------------------------------------------------------------- f32 -> f16
__global__ void convert_f2h_kernel(const float* __restrict__ src, _Float16* __restrict__ dst, int n) {
  int t = blockIdx.x * blockDim.x + threadIdx.x;
  if (t < n) dst[t] = (_Float16)src[t];
}

// ---------------------------------------------------------------- x[b][c][p] -> xT[b][p][c] (f16)
__global__ __launch_bounds__(256) void transpose_x_kernel(const float* __restrict__ x,
                                                          _Float16* __restrict__ xT) {
  __shared__ float tile[32][33];
  int b  = blockIdx.z;
  int p0 = blockIdx.x * 32;
  int c0 = blockIdx.y * 32;
  int tx = threadIdx.x, ty = threadIdx.y;
  const float* xb = x + (size_t)b * CIN * HW;
#pragma unroll
  for (int r = 0; r < 4; ++r) {
    int c = c0 + ty + r * 8;
    tile[ty + r * 8][tx] = xb[(size_t)c * HW + p0 + tx];
  }
  __syncthreads();
  _Float16* xtb = xT + (size_t)b * HW * CIN;
#pragma unroll
  for (int r = 0; r < 4; ++r) {
    int p = p0 + ty + r * 8;
    xtb[(size_t)p * CIN + c0 + tx] = (_Float16)tile[tx][ty + r * 8];
  }
}

// ---------------------------------------------------------------- adaptive pooling, all 5 scales
// one block per (b,c); output pooled[b][cell][c], cells ordered k=1,2,3,6,12
__global__ __launch_bounds__(256) void pool_kernel(const float* __restrict__ x,
                                                   float* __restrict__ pooled) {
  __shared__ float xs[4096];
  __shared__ float tmp[24 * 64];
  const int kv[5]    = {1, 2, 3, 6, 12};
  const int rpref[6] = {0, 1, 3, 6, 12, 24};
  const int cpref[6] = {0, 1, 5, 14, 50, 194};
  int b = blockIdx.x / CIN;
  int c = blockIdx.x % CIN;
  int tid = threadIdx.x;
  const float* src = x + ((size_t)b * CIN + c) * HW;
  for (int i = tid; i < 4096; i += 256) xs[i] = src[i];
  __syncthreads();
  // stage 1: row-bin sums  tmp[bin][w]
  for (int idx = tid; idx < 24 * 64; idx += 256) {
    int row = idx >> 6, w = idx & 63;
    int ki = 0; while (row >= rpref[ki + 1]) ++ki;
    int p = row - rpref[ki];
    int k = kv[ki];
    int s = (p * 64) / k, e = ((p + 1) * 64 + k - 1) / k;
    float acc = 0.f;
    for (int h = s; h < e; ++h) acc += xs[h * 64 + w];
    tmp[idx] = acc;
  }
  __syncthreads();
  // stage 2: col-bin sums -> averages
  if (tid < NCELL) {
    int ki = 0; while (tid >= cpref[ki + 1]) ++ki;
    int local = tid - cpref[ki];
    int k = kv[ki];
    int p = local / k, q = local % k;
    int sh = (p * 64) / k, eh = ((p + 1) * 64 + k - 1) / k;
    int sw = (q * 64) / k, ew = ((q + 1) * 64 + k - 1) / k;
    const float* trow = &tmp[(rpref[ki] + p) * 64];
    float acc = 0.f;
    for (int w = sw; w < ew; ++w) acc += trow[w];
    acc /= (float)((eh - sh) * (ew - sw));
    pooled[((size_t)b * NCELL + tid) * CIN + c] = acc;
  }
}

// ---------------------------------------------------------------- 1x1 conv on pooled cells (tiny N)
__global__ __launch_bounds__(256) void pool_conv_kernel(const float* __restrict__ W_psp,
                                                        const float* __restrict__ pooled,
                                                        const float* __restrict__ alpha,
                                                        const float* __restrict__ beta,
                                                        float* __restrict__ fpool) {
  __shared__ float pr[CIN];
  const int cpref[6] = {0, 1, 5, 14, 50, 194};
  int b    = blockIdx.x / NCELL;
  int cell = blockIdx.x % NCELL;
  int ki = 0; while (cell >= cpref[ki + 1]) ++ki;   // scale index 0..4
  int tid = threadIdx.x;
  const float* prow = pooled + ((size_t)b * NCELL + cell) * CIN;
  for (int i = tid; i < CIN; i += 256) pr[i] = prow[i];
  __syncthreads();
  for (int o = tid; o < CI; o += 256) {
    const float* wrow = W_psp + ((size_t)ki * CI + o) * CIN;
    float acc = 0.f;
    for (int cc = 0; cc < CIN; ++cc) acc += wrow[cc] * pr[cc];
    float a = alpha[ki * CI + o], bt = beta[ki * CI + o];
    fpool[((size_t)b * NCELL + cell) * CI + o] = fmaxf(a * acc + bt, 0.f);
  }
}

// ---------------------------------------------------------------- WMMA GEMM (double-buffered)
// D[b][p][o] = relu(alpha[o]*sum_k A[o][k]*Bsrc[b][p][k] + beta[o]);  N=4096
// Bsrc split: k < B0span -> B0 (row stride B0stride), else B1 (row stride B1stride)
// Dual output: rows ob <  Msplit -> Out0[p][ob] ; ob >= Msplit -> Out1[p][ob-Msplit]
__device__ inline v16h cat16(v8h lo, v8h hi) {
  return __builtin_shufflevector(lo, hi, 0, 1, 2, 3, 4, 5, 6, 7, 8, 9, 10, 11, 12, 13, 14, 15);
}

__global__ __launch_bounds__(256) void gemm_wmma_kernel(
    const _Float16* __restrict__ A,
    const _Float16* __restrict__ B0, int B0stride, int B0span,
    const _Float16* __restrict__ B1, int B1stride,
    int K, int Msplit,
    const float* __restrict__ alpha, const float* __restrict__ beta,
    float* __restrict__ Out0, float* __restrict__ Out1) {
  __shared__ _Float16 ldsA[2][128 * 40];   // [buf][128 rows x 32 K-halves], stride 40 (conflict-free)
  __shared__ _Float16 ldsB[2][128 * 40];
  int b  = blockIdx.z;
  int m0 = blockIdx.y * 128;
  int n0 = blockIdx.x * 128;
  const _Float16* b0p = B0 + (size_t)b * HW * B0stride;
  const _Float16* b1p = B1 + (size_t)b * HW * B1stride;
  int tid  = threadIdx.x;
  int lane = tid & 31;
  int wv   = tid >> 5;
  int wm   = wv >> 2;          // wave M group: 0..1 -> rows of 64
  int wn   = wv & 3;           // wave N group: 0..3 -> cols of 32
  int g    = lane >> 4;        // lane half-group
  int l16  = lane & 15;

  int seg = tid & 3;           // 4 x 8-half (16B) segments per row
  int row = tid >> 2;          // 0..63; rows row and row+64

  v8f acc[4][2];
  v8f zero = {0.f, 0.f, 0.f, 0.f, 0.f, 0.f, 0.f, 0.f};
#pragma unroll
  for (int mt = 0; mt < 4; ++mt)
#pragma unroll
    for (int nt = 0; nt < 2; ++nt) acc[mt][nt] = zero;

  // stage tile k0=0 into buffer 0
  {
    int kk = seg * 8;
    uint4 a0  = *(const uint4*)(A + (size_t)(m0 + row) * K + kk);
    uint4 a1  = *(const uint4*)(A + (size_t)(m0 + row + 64) * K + kk);
    const _Float16* p0 = (kk < B0span) ? b0p + (size_t)(n0 + row) * B0stride + kk
                                       : b1p + (size_t)(n0 + row) * B1stride + (kk - B0span);
    const _Float16* p1 = (kk < B0span) ? b0p + (size_t)(n0 + row + 64) * B0stride + kk
                                       : b1p + (size_t)(n0 + row + 64) * B1stride + (kk - B0span);
    uint4 bb0 = *(const uint4*)p0;
    uint4 bb1 = *(const uint4*)p1;
    *(uint4*)&ldsA[0][row * 40 + seg * 8]        = a0;
    *(uint4*)&ldsA[0][(row + 64) * 40 + seg * 8] = a1;
    *(uint4*)&ldsB[0][row * 40 + seg * 8]        = bb0;
    *(uint4*)&ldsB[0][(row + 64) * 40 + seg * 8] = bb1;
  }
  __syncthreads();

  for (int k0 = 0; k0 < K; k0 += 32) {
    int buf = (k0 >> 5) & 1;
    bool more = (k0 + 32) < K;
    uint4 a0, a1, bb0, bb1;
    if (more) {
      // issue next tile's global loads early (latency overlapped with WMMA below)
      int kk = k0 + 32 + seg * 8;
      a0 = *(const uint4*)(A + (size_t)(m0 + row) * K + kk);
      a1 = *(const uint4*)(A + (size_t)(m0 + row + 64) * K + kk);
      const _Float16* p0 = (kk < B0span) ? b0p + (size_t)(n0 + row) * B0stride + kk
                                         : b1p + (size_t)(n0 + row) * B1stride + (kk - B0span);
      const _Float16* p1 = (kk < B0span) ? b0p + (size_t)(n0 + row + 64) * B0stride + kk
                                         : b1p + (size_t)(n0 + row + 64) * B1stride + (kk - B0span);
      bb0 = *(const uint4*)p0;
      bb1 = *(const uint4*)p1;
      if (k0 + 128 <= K) {
        // speculative prefetch two tiles ahead -> global_prefetch_b8
        __builtin_prefetch(A + (size_t)(m0 + row) * K + kk + 64, 0, 1);
        int kp = kk + 64;
        const _Float16* pp = (kp < B0span) ? b0p + (size_t)(n0 + row) * B0stride + kp
                                           : b1p + (size_t)(n0 + row) * B1stride + (kp - B0span);
        __builtin_prefetch(pp, 0, 1);
      }
    }

    // compute current buffer
    v16h af[4], bf[2];
#pragma unroll
    for (int mt = 0; mt < 4; ++mt) {
      int m = wm * 64 + mt * 16 + l16;
      // A 16x32 f16 layout: lane group g: v0-3 <- K = g*8+0..7 ; v4-7 <- K = 16+g*8+0..7
      v8h lo = *(const v8h*)&ldsA[buf][m * 40 + g * 8];
      v8h hi = *(const v8h*)&ldsA[buf][m * 40 + 16 + g * 8];
      af[mt] = cat16(lo, hi);
    }
#pragma unroll
    for (int nt = 0; nt < 2; ++nt) {
      int n = wn * 32 + nt * 16 + l16;
      // B 32x16 f16 layout: lane group g: v0-3 <- K = g*8+0..7 ; v4-7 <- K = 16+g*8+0..7
      v8h lo = *(const v8h*)&ldsB[buf][n * 40 + g * 8];
      v8h hi = *(const v8h*)&ldsB[buf][n * 40 + 16 + g * 8];
      bf[nt] = cat16(lo, hi);
    }
#pragma unroll
    for (int mt = 0; mt < 4; ++mt)
#pragma unroll
      for (int nt = 0; nt < 2; ++nt)
        acc[mt][nt] = __builtin_amdgcn_wmma_f32_16x16x32_f16(
            false, af[mt], false, bf[nt], (short)0, acc[mt][nt], false, false);

    if (more) {
      int nb = buf ^ 1;
      *(uint4*)&ldsA[nb][row * 40 + seg * 8]        = a0;
      *(uint4*)&ldsA[nb][(row + 64) * 40 + seg * 8] = a1;
      *(uint4*)&ldsB[nb][row * 40 + seg * 8]        = bb0;
      *(uint4*)&ldsB[nb][(row + 64) * 40 + seg * 8] = bb1;
    }
    __syncthreads();
  }

  // epilogue: D vgpr r -> M = r + 8*g, N = l16;  store p-major [p][o]
#pragma unroll
  for (int mt = 0; mt < 4; ++mt) {
    int ob = m0 + wm * 64 + mt * 16 + g * 8;
    float* outp;
    int oo;
    if (ob < Msplit) { outp = Out0; oo = ob; }
    else             { outp = Out1; oo = ob - Msplit; }
#pragma unroll
    for (int nt = 0; nt < 2; ++nt) {
      int p = n0 + wn * 32 + nt * 16 + l16;
      float* dst = outp + ((size_t)b * HW + p) * CI + oo;
      float4 v0, v1;
      v0.x = fmaxf(alpha[ob + 0] * acc[mt][nt][0] + beta[ob + 0], 0.f);
      v0.y = fmaxf(alpha[ob + 1] * acc[mt][nt][1] + beta[ob + 1], 0.f);
      v0.z = fmaxf(alpha[ob + 2] * acc[mt][nt][2] + beta[ob + 2], 0.f);
      v0.w = fmaxf(alpha[ob + 3] * acc[mt][nt][3] + beta[ob + 3], 0.f);
      v1.x = fmaxf(alpha[ob + 4] * acc[mt][nt][4] + beta[ob + 4], 0.f);
      v1.y = fmaxf(alpha[ob + 5] * acc[mt][nt][5] + beta[ob + 5], 0.f);
      v1.z = fmaxf(alpha[ob + 6] * acc[mt][nt][6] + beta[ob + 6], 0.f);
      v1.w = fmaxf(alpha[ob + 7] * acc[mt][nt][7] + beta[ob + 7], 0.f);
      *(float4*)dst       = v0;
      *(float4*)(dst + 4) = v1;
    }
  }
}

// ---------------------------------------------------------------- attention (one wave32 per pixel)
__global__ __launch_bounds__(256) void attention_kernel(const float* __restrict__ guideT,
                                                        const float* __restrict__ f5T,
                                                        const float* __restrict__ fpool,
                                                        const float* __restrict__ gammaPtr,
                                                        _Float16* __restrict__ psaa3T) {
  const int kv[5]    = {1, 2, 3, 6, 12};
  const int cbase[5] = {0, 1, 5, 14, 50};
  int lane = threadIdx.x & 31;
  int wv   = threadIdx.x >> 5;
  int pix  = blockIdx.x * 8 + wv;
  int b = pix >> 12;
  int p = pix & 4095;
  int h = p >> 6, w = p & 63;
  float gamma = gammaPtr[0];
  const float* gRow = guideT + (size_t)pix * CI;
  const float* fRow = f5T + (size_t)pix * CI;
  float gch[16], y[6][16];
#pragma unroll
  for (int j = 0; j < 16; ++j) {
    int cc = j * 32 + lane;
    gch[j]  = gRow[cc];
    y[5][j] = fRow[cc];
  }
  const float* fb = fpool + (size_t)b * NCELL * CI;
#pragma unroll
  for (int s = 0; s < 5; ++s) {
    int k = kv[s];
    float sc = (k > 1) ? (float)(k - 1) / 63.0f : 0.0f;
    float ry = h * sc; int y0 = (int)ry; int y1 = min(y0 + 1, k - 1); float fy = ry - y0;
    float rx = w * sc; int x0 = (int)rx; int x1 = min(x0 + 1, k - 1); float fx = rx - x0;
    const float* r00 = fb + (size_t)(cbase[s] + y0 * k + x0) * CI;
    const float* r01 = fb + (size_t)(cbase[s] + y0 * k + x1) * CI;
    const float* r10 = fb + (size_t)(cbase[s] + y1 * k + x0) * CI;
    const float* r11 = fb + (size_t)(cbase[s] + y1 * k + x1) * CI;
    float w00 = (1 - fy) * (1 - fx), w01 = (1 - fy) * fx, w10 = fy * (1 - fx), w11 = fy * fx;
#pragma unroll
    for (int j = 0; j < 16; ++j) {
      int cc = j * 32 + lane;
      y[s][j] = w00 * r00[cc] + w01 * r01[cc] + w10 * r10[cc] + w11 * r11[cc];
    }
  }
  float e[6];
#pragma unroll
  for (int s = 0; s < 6; ++s) {
    float part = 0.f;
#pragma unroll
    for (int j = 0; j < 16; ++j) part += gch[j] * y[s][j];
#pragma unroll
    for (int off = 16; off > 0; off >>= 1) part += __shfl_xor(part, off, 32);
    e[s] = part;
  }
  float mx = e[0];
#pragma unroll
  for (int s = 1; s < 6; ++s) mx = fmaxf(mx, e[s]);
  float den = 0.f, att[6];
#pragma unroll
  for (int s = 0; s < 6; ++s) { att[s] = __expf(e[s] - mx); den += att[s]; }
  float inv = 1.0f / den;
  _Float16* out = psaa3T + (size_t)pix * CI;
#pragma unroll
  for (int j = 0; j < 16; ++j) {
    float acc = 0.f;
#pragma unroll
    for (int s = 0; s < 6; ++s) acc += att[s] * y[s][j];
    out[j * 32 + lane] = (_Float16)(gamma * acc * inv + gch[j]);
  }
}

// ---------------------------------------------------------------- 19-class heads (one wave per pixel)
__global__ __launch_bounds__(256) void heads_kernel(const float* __restrict__ fsumT,
                                                    const float* __restrict__ guideT,
                                                    const float* __restrict__ W8, const float* __restrict__ b8,
                                                    const float* __restrict__ Wg, const float* __restrict__ bg,
                                                    float* __restrict__ out) {
  int lane = threadIdx.x & 31;
  int wv   = threadIdx.x >> 5;
  int pix  = blockIdx.x * 8 + wv;
  int b = pix >> 12, p = pix & 4095;
  const float* f  = fsumT + (size_t)pix * CI;
  const float* gd = guideT + (size_t)pix * CI;
  float fr[16], gr[16];
#pragma unroll
  for (int j = 0; j < 16; ++j) { int cc = j * 32 + lane; fr[j] = f[cc]; gr[j] = gd[cc]; }
  float* out8 = out;
  float* gp   = out + (size_t)NB * NCLS * HW;
  for (int o = 0; o < NCLS; ++o) {
    float s8 = 0.f, sg = 0.f;
    const float* w8r = W8 + o * CI;
    const float* wgr = Wg + o * CI;
#pragma unroll
    for (int j = 0; j < 16; ++j) {
      int cc = j * 32 + lane;
      s8 += w8r[cc] * fr[j];
      sg += wgr[cc] * gr[j];
    }
#pragma unroll
    for (int off = 16; off > 0; off >>= 1) {
      s8 += __shfl_xor(s8, off, 32);
      sg += __shfl_xor(sg, off, 32);
    }
    if (lane == 0) {
      out8[((size_t)b * NCLS + o) * HW + p] = s8 + b8[o];
      gp  [((size_t)b * NCLS + o) * HW + p] = sg + bg[o];
    }
  }
}

// ================================================================ host launcher
extern "C" void kernel_launch(void* const* d_in, const int* in_sizes, int n_in,
                              void* d_out, int out_size, void* d_ws, size_t ws_size,
                              hipStream_t stream) {
  const float* x     = (const float*)d_in[0];
  const float* W_psp = (const float*)d_in[1];
  const float* bn_s  = (const float*)d_in[2];
  const float* bn_b  = (const float*)d_in[3];
  const float* bn_m  = (const float*)d_in[4];
  const float* bn_v  = (const float*)d_in[5];
  const float* gamma = (const float*)d_in[6];
  const float* W52   = (const float*)d_in[7];
  const float* s52   = (const float*)d_in[8];
  const float* b52   = (const float*)d_in[9];
  const float* m52   = (const float*)d_in[10];
  const float* v52   = (const float*)d_in[11];
  const float* Wg    = (const float*)d_in[12];
  const float* bg    = (const float*)d_in[13];
  const float* W8    = (const float*)d_in[14];
  const float* b8    = (const float*)d_in[15];

  uint8_t* ws = (uint8_t*)d_ws;
  size_t off = 0;
  auto take = [&](size_t bytes) -> void* {
    void* ptr = ws + off;
    off += (bytes + 255) & ~(size_t)255;
    return ptr;
  };
  _Float16* xT      = (_Float16*)take((size_t)NB * HW * CIN * 2);        // 33.6 MB
  _Float16* Wh56    = (_Float16*)take((size_t)2 * CI * CIN * 2);         //  4.2 MB (i=5 then i=6)
  _Float16* W52h    = (_Float16*)take((size_t)CI * 2560 * 2);            //  2.6 MB
  float*    alpha   = (float*)take((size_t)8 * CI * 4);
  float*    beta    = (float*)take((size_t)8 * CI * 4);
  float*    pooled  = (float*)take((size_t)NB * NCELL * CIN * 4);        //  3.2 MB
  float*    fpool   = (float*)take((size_t)NB * NCELL * CI * 4);         //  0.8 MB
  float*    f5T     = (float*)take((size_t)NB * HW * CI * 4);            // 16.8 MB
  float*    guideT  = (float*)take((size_t)NB * HW * CI * 4);            // 16.8 MB
  _Float16* psaa3T  = (_Float16*)take((size_t)NB * HW * CI * 2);         //  8.4 MB
  float*    fsumT   = (float*)take((size_t)NB * HW * CI * 4);            // 16.8 MB
  (void)ws_size; (void)in_sizes; (void)n_in; (void)out_size;

  // 1) fold BN params into alpha/beta (8 sets: 7 PSP convs + conv52)
  fold_ab_kernel<<<(8 * CI + 255) / 256, 256, 0, stream>>>(
      bn_s, bn_b, bn_m, bn_v, s52, b52, m52, v52, alpha, beta);

  // 2) f16 weights for the big GEMMs
  convert_f2h_kernel<<<(2 * CI * CIN + 255) / 256, 256, 0, stream>>>(
      W_psp + (size_t)5 * CI * CIN, Wh56, 2 * CI * CIN);
  convert_f2h_kernel<<<(CI * 2560 + 255) / 256, 256, 0, stream>>>(W52, W52h, CI * 2560);

  // 3) x -> xT (p-major f16, K contiguous)
  transpose_x_kernel<<<dim3(128, 64, NB), dim3(32, 8), 0, stream>>>(x, xT);

  // 4) adaptive pooling (all 5 scales)
  pool_kernel<<<NB * CIN, 256, 0, stream>>>(x, pooled);

  // 5) 1x1 conv + BN + ReLU on pooled cells
  pool_conv_kernel<<<NB * NCELL, 256, 0, stream>>>(W_psp, pooled, alpha, beta, fpool);

  // 6) fused conv5+conv6 (WMMA, M=1024, K=2048): rows<512 -> f5T, rows>=512 -> guideT
  gemm_wmma_kernel<<<dim3(32, 8, NB), 256, 0, stream>>>(
      Wh56, xT, CIN, CIN, xT, CIN, CIN, CI,
      alpha + 5 * CI, beta + 5 * CI, f5T, guideT);

  // 7) scale attention + psaa3 (f16, p-major)
  attention_kernel<<<NB * HW / 8, 256, 0, stream>>>(guideT, f5T, fpool, gamma, psaa3T);

  // 8) conv52 over concat([psaa3, x])  (WMMA, K=2560, split B source)
  gemm_wmma_kernel<<<dim3(32, 4, NB), 256, 0, stream>>>(
      W52h, psaa3T, CI, CI, xT, CIN, 2560, CI,
      alpha + 7 * CI, beta + 7 * CI, fsumT, fsumT);

  // 9) classification heads -> d_out (out8 then guide_pred)
  heads_kernel<<<NB * HW / 8, 256, 0, stream>>>(fsumT, guideT, W8, b8, Wg, bg, (float*)d_out);
}
